// CholecFixScore_8160437862601
// MI455X (gfx1250) — 8-bit matrix operand order repaired and checked by reading, compiled, not run on hardware
//
#include <hip/hip_runtime.h>
#include <hip/hip_bf16.h>


typedef __attribute__((ext_vector_type(8))) int v8i;

#define N_ 16
#define P_ 16
#define T_ 8
#define HW_ (224 * 224)
#define CHUNKS1 49
#define TILES_PER_CHUNK 16
#define CH3 49

__device__ __forceinline__ unsigned pack4(float4 v) {
    return (v.x != 0.0f ? 0x00000001u : 0u) |
           (v.y != 0.0f ? 0x00000100u : 0u) |
           (v.z != 0.0f ? 0x00010000u : 0u) |
           (v.w != 0.0f ? 0x01000000u : 0u);
}

__global__ __launch_bounds__(256) void zero_ws_kernel(int* __restrict__ inters,
                                                      int* __restrict__ st) {
    int i = blockIdx.x * 256 + threadIdx.x;
    if (i < N_ * 16 * 16) inters[i] = 0;
    if (i < N_ * T_) st[i] = 0;
}

__global__ __launch_bounds__(32) void phase1_wmma_kernel(
        const float* __restrict__ gp, const float* __restrict__ gt,
        int* __restrict__ inters, int* __restrict__ st) {
    const int n    = blockIdx.x / CHUNKS1;
    const int c    = blockIdx.x % CHUNKS1;
    const int lane = threadIdx.x;
    const int half = lane >> 4;
    const int pr   = lane & 15;
    const int tc   = lane & 15;

    const float* gprow = gp + (size_t)(n * P_ + pr) * HW_;
    const float* gtrow = gt + (size_t)(n * T_ + (tc < T_ ? tc : 0)) * HW_;

    v8i cacc = {0, 0, 0, 0, 0, 0, 0, 0};
    int stcnt = 0;

    const int kchunk = c * (TILES_PER_CHUNK * 64);
    for (int t = 0; t < TILES_PER_CHUNK; ++t) {
        const int kbase = kchunk + t * 64;

        v8i A;
#pragma unroll
        for (int g = 0; g < 4; ++g) {
            const float4* src =
                reinterpret_cast<const float4*>(gprow + kbase + (g >> 1) * 32 + (g & 1) * 8 + half * 16);
            A[2 * g]     = (int)pack4(src[0]);
            A[2 * g + 1] = (int)pack4(src[1]);
        }

        v8i B;
        if (tc < T_) {
            const float4* s0 =
                reinterpret_cast<const float4*>(gtrow + kbase + half * 16);
            const float4* s1 =
                reinterpret_cast<const float4*>(gtrow + kbase + 32 + half * 16);
#pragma unroll
            for (int v = 0; v < 4; ++v) {
                unsigned p0 = pack4(s0[v]);
                unsigned p1 = pack4(s1[v]);
                B[v]     = (int)p0;
                B[v + 4] = (int)p1;
                stcnt += __popc(p0) + __popc(p1);
            }
        } else if (tc == T_) {
#pragma unroll
            for (int v = 0; v < 8; ++v) B[v] = 0x01010101;
        } else {
#pragma unroll
            for (int v = 0; v < 8; ++v) B[v] = 0;
        }

        cacc = __builtin_amdgcn_wmma_i32_16x16x64_iu8(
             false, A,  false, B, cacc,
             false,  false);
    }

    if (tc <= T_) {
#pragma unroll
        for (int r = 0; r < 8; ++r) {
            const int M = r + 8 * half;
            atomicAdd(&inters[(n * P_ + M) * 16 + tc], cacc[r]);
        }
    }
    if (tc < T_) atomicAdd(&st[n * T_ + tc], stcnt);
}

__global__ __launch_bounds__(256) void phase2_ioumax_kernel(
        const int* __restrict__ inters, const int* __restrict__ st,
        float* __restrict__ ioumax) {
    const int idx = threadIdx.x;
    const int n = idx >> 4, p = idx & 15;
    const int* row = &inters[(n * P_ + p) * 16];
    const float sp = (float)row[T_];
    float best = 0.0f;
#pragma unroll
    for (int t = 0; t < T_; ++t) {
        const float inter = (float)row[t];
        const float u = sp + (float)st[n * T_ + t] - inter;
        const float iou = (u > 0.0f) ? inter / u : 0.0f;
        best = fmaxf(best, iou);
    }
    ioumax[idx] = best;
}

__global__ __launch_bounds__(256) void phase3_score_kernel(
        const float* __restrict__ gp, const float* __restrict__ ioumax,
        float* __restrict__ partials) {
    const int n = blockIdx.x / CH3;
    const int c = blockIdx.x % CH3;
    __shared__ float w[P_];
    __shared__ float red[256];
    if (threadIdx.x < P_) w[threadIdx.x] = ioumax[n * P_ + threadIdx.x];
    __syncthreads();

    const float* base = gp + (size_t)n * P_ * HW_;
    const int hw0 = c * 1024 + threadIdx.x;
    float acc = 0.0f;
#pragma unroll
    for (int k = 0; k < 4; ++k) {
        const int hw = hw0 + k * 256;
        float s = 0.0f;
        int cnt = 0;
#pragma unroll
        for (int p = 0; p < P_; ++p) {
            const bool b = (base[(size_t)p * HW_ + hw] != 0.0f);
            s += b ? w[p] : 0.0f;
            cnt += b ? 1 : 0;
        }
        acc += (cnt > 0) ? s / (float)cnt : 0.0f;
    }

    red[threadIdx.x] = acc;
    __syncthreads();
    for (int off = 128; off > 0; off >>= 1) {
        if (threadIdx.x < off) red[threadIdx.x] += red[threadIdx.x + off];
        __syncthreads();
    }
    if (threadIdx.x == 0) partials[blockIdx.x] = red[0];
}

__global__ __launch_bounds__(32) void phase4_finalize_kernel(
        const float* __restrict__ partials, float* __restrict__ out) {
    const int n = threadIdx.x;
    if (n < N_) {
        float s = 0.0f;
        for (int c = 0; c < CH3; ++c) s += partials[n * CH3 + c];
        out[n] = s / (float)HW_;
    }
}

extern "C" void kernel_launch(void* const* d_in, const int* in_sizes, int n_in,
                              void* d_out, int out_size, void* d_ws, size_t ws_size,
                              hipStream_t stream) {
    const float* gp = (const float*)d_in[0];
    const float* gt = (const float*)d_in[1];
    float* out = (float*)d_out;

    char* ws = (char*)d_ws;
    int*   inters   = (int*)(ws);
    int*   st       = (int*)(ws + 16384);
    float* ioumax   = (float*)(ws + 16384 + 512);
    float* partials = (float*)(ws + 16384 + 512 + 1024);

    zero_ws_kernel<<<16, 256, 0, stream>>>(inters, st);
    phase1_wmma_kernel<<<N_ * CHUNKS1, 32, 0, stream>>>(gp, gt, inters, st);
    phase2_ioumax_kernel<<<1, 256, 0, stream>>>(inters, st, ioumax);
    phase3_score_kernel<<<N_ * CH3, 256, 0, stream>>>(gp, ioumax, partials);
    phase4_finalize_kernel<<<1, 32, 0, stream>>>(partials, out);
}
